// SSGCN_10196252361189
// MI455X (gfx1250) — matrix operand order repaired and checked by reading, not compiled, not run on hardware
//
#include <hip/hip_runtime.h>
#include <hip/hip_bf16.h>

// ---------------- problem constants (from reference) ----------------
#define BATCH  128
#define S_LEN  1024
#define NEDGE  2048
#define H_DIM  300
#define NASP   5
#define NCLS   3

// ---------------- GEMM tiling ----------------
#define MT 128      // M tile (rows of X)
#define NT 160      // N tile: 10 x 16 subtiles, 2 N-passes cover 300 (pad 320)
#define NSUB 10     // N subtiles per wave
#define KT 32       // K step = one v_wmma_f32_16x16x32_bf16
#define KITERS 10   // ceil(300/32) padded with zeros to 320
#define A_STRIDE 40   // ushorts per A row in LDS (32 + pad, keeps 16B align)
#define B_STRIDE 168  // ushorts per B row in LDS (160 + pad, keeps 16B align)

typedef __attribute__((ext_vector_type(16))) __bf16 v16bf;
typedef __attribute__((ext_vector_type(8)))  float  v8f;

union Frag16 {            // 8 VGPRs viewed as 16 bf16 or 2x 16-byte chunks
  v16bf v;
  uint4 q[2];
};

static __device__ __forceinline__ unsigned short f2bf(float f) {
  __bf16 b = (__bf16)f;                      // v_cvt_pk_bf16_f32 when paired
  return __builtin_bit_cast(unsigned short, b);
}

// =====================================================================
// Kernel 1: aspect mean + q = avec @ W_span + b_span   (one block/sample)
// =====================================================================
__global__ __launch_bounds__(256)
void aspectq_kernel(const float* __restrict__ h, const int* __restrict__ ap,
                    const float* __restrict__ W_span, const float* __restrict__ b_span,
                    float* __restrict__ q) {
  const int b = blockIdx.x;
  __shared__ float avec[H_DIM];
  __shared__ int   pos[NASP];
  if (threadIdx.x < NASP) pos[threadIdx.x] = ap[b * NASP + threadIdx.x];
  __syncthreads();
  for (int j = threadIdx.x; j < H_DIM; j += 256) {
    float s = 0.f;
    #pragma unroll
    for (int a = 0; a < NASP; ++a)
      s += h[((size_t)b * S_LEN + pos[a]) * H_DIM + j];
    avec[j] = s * (1.f / NASP);
  }
  __syncthreads();
  for (int j = threadIdx.x; j < H_DIM; j += 256) {
    float s = b_span[j];
    for (int k = 0; k < H_DIM; ++k)
      s += avec[k] * W_span[(size_t)k * H_DIM + j];
    q[(size_t)b * H_DIM + j] = s;
  }
}

// =====================================================================
// Kernel 2: attn = softmax(h_i @ q); span = attn @ h_i  (one block/sample)
// =====================================================================
__global__ __launch_bounds__(256)
void span_kernel(const float* __restrict__ h, const float* __restrict__ q,
                 float* __restrict__ span) {
  const int b = blockIdx.x;
  const int t = threadIdx.x, lane = t & 31, wv = t >> 5;
  __shared__ float qs[H_DIM];
  __shared__ float attn[S_LEN];
  __shared__ float red[8];
  for (int j = t; j < H_DIM; j += 256) qs[j] = q[(size_t)b * H_DIM + j];
  __syncthreads();
  const float* hb = h + (size_t)b * S_LEN * H_DIM;
  // logits: one wave per row, lanes split K
  for (int s = wv; s < S_LEN; s += 8) {
    float d = 0.f;
    for (int k = lane; k < H_DIM; k += 32) d += hb[(size_t)s * H_DIM + k] * qs[k];
    #pragma unroll
    for (int off = 16; off > 0; off >>= 1) d += __shfl_xor(d, off, 32);
    if (lane == 0) attn[s] = d;
  }
  __syncthreads();
  // softmax over S_LEN in LDS
  float m = -1e30f;
  for (int s = t; s < S_LEN; s += 256) m = fmaxf(m, attn[s]);
  #pragma unroll
  for (int off = 16; off > 0; off >>= 1) m = fmaxf(m, __shfl_xor(m, off, 32));
  if (lane == 0) red[wv] = m;
  __syncthreads();
  float mm = red[0];
  #pragma unroll
  for (int i = 1; i < 8; ++i) mm = fmaxf(mm, red[i]);
  __syncthreads();   // everyone has read red before it is reused
  float sum = 0.f;
  for (int s = t; s < S_LEN; s += 256) {
    float e = __expf(attn[s] - mm);
    attn[s] = e;
    sum += e;
  }
  #pragma unroll
  for (int off = 16; off > 0; off >>= 1) sum += __shfl_xor(sum, off, 32);
  if (lane == 0) red[wv] = sum;
  __syncthreads();
  float tot = 0.f;
  #pragma unroll
  for (int i = 0; i < 8; ++i) tot += red[i];
  const float inv = 1.f / tot;
  // span[j] = sum_s attn[s] * h[s][j]  (threads over j -> coalesced h reads)
  for (int j = t; j < H_DIM; j += 256) {
    float acc = 0.f;
    for (int s = 0; s < S_LEN; ++s) acc += attn[s] * hb[(size_t)s * H_DIM + j];
    span[(size_t)b * H_DIM + j] = acc * inv;
  }
}

// =====================================================================
// Kernel 3: degree (weighted in-degree + 1 self loop), LDS float atomics
// =====================================================================
__global__ __launch_bounds__(256)
void deg_kernel(const int* __restrict__ ei, const float* __restrict__ ew,
                float* __restrict__ deg) {
  const int b = blockIdx.x;
  __shared__ float dl[S_LEN];
  for (int s = threadIdx.x; s < S_LEN; s += 256) dl[s] = 0.f;
  __syncthreads();
  const int*   dst = ei + ((size_t)b * 2 + 1) * NEDGE;
  const float* ewb = ew + (size_t)b * NEDGE;
  for (int e = threadIdx.x; e < NEDGE; e += 256)
    atomicAdd(&dl[dst[e]], ewb[e]);               // ds_add_f32
  __syncthreads();
  for (int s = threadIdx.x; s < S_LEN; s += 256)
    deg[(size_t)b * S_LEN + s] = dl[s] + 1.0f;
}

// =====================================================================
// Kernel 4: Out[M x 300] = X[M x 300] @ W[300 x 300], bf16 WMMA, f32 acc
//   block = 256 thr = 8 waves; tile 128(M) x 160(N); K padded 300 -> 320
//   fp32 -> bf16 converted ONCE at LDS stage time (halves LDS bytes,
//   removes per-consumer cvt); A frag = two ds_load_b128, B frag = 16 strided u16 loads.
//   wave w computes rows [16w,16w+16) x all 10 N-subtiles
// =====================================================================
__global__ __launch_bounds__(256)
void gemm_bf16wmma_kernel(const float* __restrict__ X, const float* __restrict__ W,
                          float* __restrict__ Out) {
  __shared__ __align__(16) unsigned short As[MT][A_STRIDE];   // 128x32 bf16 (+pad)
  __shared__ __align__(16) unsigned short Bs[KT][B_STRIDE];   // 32x160 bf16 (+pad)
  const int t    = threadIdx.x;
  const int wv   = t >> 5;
  const int lane = t & 31;
  const int half = lane >> 4;        // lane group (0: lanes 0-15, 1: 16-31)
  const int m0   = blockIdx.x * MT;
  const int n0   = blockIdx.y * NT;

  v8f acc[NSUB] = {};

  for (int kk = 0; kk < KITERS; ++kk) {
    const int k0 = kk * KT;
    // ---- stage A tile: 128x32, fp32 load -> bf16 LDS (4 float4 / thread) ----
    for (int i = t; i < (MT * KT) / 4; i += 256) {
      const int row = i >> 3;           // 8 float4 per row
      const int c4  = (i & 7) * 4;
      const int gk  = k0 + c4;
      const float* src = X + (size_t)(m0 + row) * H_DIM + gk;
      float4 v;
      if (gk + 3 < H_DIM) {
        v = *(const float4*)src;
      } else {
        v.x = (gk + 0 < H_DIM) ? src[0] : 0.f;
        v.y = (gk + 1 < H_DIM) ? src[1] : 0.f;
        v.z = (gk + 2 < H_DIM) ? src[2] : 0.f;
        v.w = (gk + 3 < H_DIM) ? src[3] : 0.f;
      }
      ushort4 s4;
      s4.x = f2bf(v.x); s4.y = f2bf(v.y); s4.z = f2bf(v.z); s4.w = f2bf(v.w);
      *(ushort4*)&As[row][c4] = s4;     // 8B aligned (c4 multiple of 4)
    }
    // ---- stage B tile: 32x160, fp32 load -> bf16 LDS (5 float4 / thread) ----
    for (int i = t; i < (KT * NT) / 4; i += 256) {
      const int krow = i / (NT / 4);    // 40 float4 per row
      const int c4   = (i % (NT / 4)) * 4;
      const int gk   = k0 + krow;
      const int gn   = n0 + c4;
      float4 v = {0.f, 0.f, 0.f, 0.f};
      if (gk < H_DIM) {
        const float* src = W + (size_t)gk * H_DIM + gn;
        if (gn + 3 < H_DIM) {
          v = *(const float4*)src;
        } else {
          v.x = (gn + 0 < H_DIM) ? src[0] : 0.f;
          v.y = (gn + 1 < H_DIM) ? src[1] : 0.f;
          v.z = (gn + 2 < H_DIM) ? src[2] : 0.f;
          v.w = (gn + 3 < H_DIM) ? src[3] : 0.f;
        }
      }
      ushort4 s4;
      s4.x = f2bf(v.x); s4.y = f2bf(v.y); s4.z = f2bf(v.z); s4.w = f2bf(v.w);
      *(ushort4*)&Bs[krow][c4] = s4;
    }
    // prefetch next K-tile of A into cache (global_prefetch_b8)
    if (kk + 1 < KITERS)
      __builtin_prefetch(X + (size_t)(m0 + (t >> 1)) * H_DIM + (kk + 1) * KT, 0, 1);
    __syncthreads();

    // ---- A fragment: 16x32 bf16, ISA layout:
    //   lanes 0-15 row M=lane, elems 0-7 -> K 0-7,  8-15 -> K 16-23
    //   lanes 16-31 row M=lane-16, elems 0-7 -> K 8-15, 8-15 -> K 24-31
    Frag16 a;
    const unsigned short* ar = &As[(wv << 4) + (lane & 15)][0];
    a.q[0] = *(const uint4*)&ar[half << 3];          // K = 8*half .. +7
    a.q[1] = *(const uint4*)&ar[16 + (half << 3)];   // K = 16+8*half .. +7
    // ---- B fragments: 32x16 bf16, lane -> N = lane&15, elems 0-7 -> K 8*half.., 8-15 -> K 16+8*half.. ----
    #pragma unroll
    for (int j = 0; j < NSUB; ++j) {
      Frag16 bb;
      const unsigned short* br = &Bs[0][(j << 4) + (lane & 15)];
      #pragma unroll
      for (int s = 0; s < 8; ++s) {
        bb.v[s]     = __builtin_bit_cast(__bf16, br[((half << 3) + s) * B_STRIDE]);
        bb.v[8 + s] = __builtin_bit_cast(__bf16, br[(16 + (half << 3) + s) * B_STRIDE]);
      }
      acc[j] = __builtin_amdgcn_wmma_f32_16x16x32_bf16(
          /*neg_a=*/false, a.v, /*neg_b=*/false, bb.v,
          /*c_mod=*/(short)0, acc[j], /*reuse_a=*/false, /*reuse_b=*/false);
    }
    __syncthreads();
  }

  // ---- store D: lane -> N = lane&15, VGPR r -> M = r + 8*half ----
  const int rowbase = m0 + (wv << 4) + (half << 3);
  const int colbase = n0 + (lane & 15);
  #pragma unroll
  for (int j = 0; j < NSUB; ++j) {
    const int col = colbase + (j << 4);
    if (col < H_DIM) {
      #pragma unroll
      for (int r = 0; r < 8; ++r)
        Out[(size_t)(rowbase + r) * H_DIM + col] = acc[j][r];
    }
  }
}

// =====================================================================
// Kernel 5: out = xw / deg + bias   (self-loop term + bias, full init)
// =====================================================================
__global__ __launch_bounds__(256)
void combine_init_kernel(const float* __restrict__ xw, const float* __restrict__ deg,
                         const float* __restrict__ bias, float* __restrict__ out,
                         int total) {
  const int i = blockIdx.x * 256 + threadIdx.x;
  if (i < total) {
    const int ch  = i % H_DIM;
    const int row = i / H_DIM;
    out[i] = xw[i] / deg[row] + bias[ch];
  }
}

// =====================================================================
// Kernel 6: scatter messages: out[dst] += dinv[src]*ew*dinv[dst]*xw[src]
//   one wave per edge, lanes split channels; global_atomic_add_f32
// =====================================================================
__global__ __launch_bounds__(256)
void scatter_kernel(const float* __restrict__ xw, const float* __restrict__ deg,
                    const int* __restrict__ ei, const float* __restrict__ ew,
                    float* __restrict__ out) {
  const int b    = blockIdx.y;
  const int lane = threadIdx.x & 31;
  const int wv   = threadIdx.x >> 5;
  const int e    = blockIdx.x * 8 + wv;   // gridDim.x = NEDGE/8
  const int* srcp = ei + (size_t)b * 2 * NEDGE;
  const int* dstp = srcp + NEDGE;
  const int s = srcp[e];
  const int d = dstp[e];
  const float w    = ew[(size_t)b * NEDGE + e];
  const float norm = rsqrtf(deg[(size_t)b * S_LEN + s]) * w *
                     rsqrtf(deg[(size_t)b * S_LEN + d]);
  const float* xs = xw  + ((size_t)b * S_LEN + s) * H_DIM;
  float*       od = out + ((size_t)b * S_LEN + d) * H_DIM;
  for (int c = lane; c < H_DIM; c += 32)
    atomicAdd(&od[c], norm * xs[c]);
}

// =====================================================================
// Kernel 7: elementwise relu
// =====================================================================
__global__ __launch_bounds__(256)
void relu_kernel(float* __restrict__ x, int total) {
  const int i = blockIdx.x * 256 + threadIdx.x;
  if (i < total) x[i] = fmaxf(x[i], 0.f);
}

// =====================================================================
// Kernel 8: syn = mean over aspect positions of x2
// =====================================================================
__global__ __launch_bounds__(256)
void syn_kernel(const float* __restrict__ x, const int* __restrict__ ap,
                float* __restrict__ syn) {
  const int b = blockIdx.x;
  __shared__ int pos[NASP];
  if (threadIdx.x < NASP) pos[threadIdx.x] = ap[b * NASP + threadIdx.x];
  __syncthreads();
  for (int j = threadIdx.x; j < H_DIM; j += 256) {
    float s = 0.f;
    #pragma unroll
    for (int a = 0; a < NASP; ++a)
      s += x[((size_t)b * S_LEN + pos[a]) * H_DIM + j];
    syn[(size_t)b * H_DIM + j] = s * (1.f / NASP);
  }
}

// =====================================================================
// Kernel 9: gate = sigmoid(cat(span,syn)@Wg + bg); fused; out = fused@Wc + bc
// =====================================================================
__global__ __launch_bounds__(256)
void final_kernel(const float* __restrict__ span, const float* __restrict__ syn,
                  const float* __restrict__ Wg, const float* __restrict__ bg,
                  const float* __restrict__ Wc, const float* __restrict__ bc,
                  float* __restrict__ out) {
  const int b = blockIdx.x;
  const int t = threadIdx.x;
  __shared__ float cat[2 * H_DIM];
  __shared__ float fused[H_DIM];
  __shared__ float red[NCLS][256];
  for (int j = t; j < H_DIM; j += 256) {
    cat[j]         = span[(size_t)b * H_DIM + j];
    cat[H_DIM + j] = syn [(size_t)b * H_DIM + j];
  }
  __syncthreads();
  for (int j = t; j < H_DIM; j += 256) {
    float s = bg[j];
    for (int k = 0; k < 2 * H_DIM; ++k)
      s += cat[k] * Wg[(size_t)k * H_DIM + j];
    const float g = 1.f / (1.f + __expf(-s));
    fused[j] = g * cat[H_DIM + j] + (1.f - g) * cat[j];
  }
  __syncthreads();
  float p[NCLS] = {0.f, 0.f, 0.f};
  for (int j = t; j < H_DIM; j += 256) {
    const float f = fused[j];
    #pragma unroll
    for (int c = 0; c < NCLS; ++c) p[c] += f * Wc[(size_t)j * NCLS + c];
  }
  #pragma unroll
  for (int c = 0; c < NCLS; ++c) red[c][t] = p[c];
  __syncthreads();
  for (int off = 128; off > 0; off >>= 1) {
    if (t < off) {
      #pragma unroll
      for (int c = 0; c < NCLS; ++c) red[c][t] += red[c][t + off];
    }
    __syncthreads();
  }
  if (t == 0) {
    #pragma unroll
    for (int c = 0; c < NCLS; ++c)
      out[(size_t)b * NCLS + c] = red[c][0] + bc[c];
  }
}

// =====================================================================
extern "C" void kernel_launch(void* const* d_in, const int* in_sizes, int n_in,
                              void* d_out, int out_size, void* d_ws, size_t ws_size,
                              hipStream_t stream) {
  const float* h      = (const float*)d_in[0];
  const int*   ei     = (const int*)  d_in[1];
  const float* ew     = (const float*)d_in[2];
  const int*   ap     = (const int*)  d_in[3];
  const float* W_span = (const float*)d_in[4];
  const float* b_span = (const float*)d_in[5];
  const float* W1     = (const float*)d_in[6];
  const float* b1     = (const float*)d_in[7];
  const float* W2     = (const float*)d_in[8];
  const float* b2     = (const float*)d_in[9];
  const float* Wg     = (const float*)d_in[10];
  const float* bg     = (const float*)d_in[11];
  const float* Wc     = (const float*)d_in[12];
  const float* bc     = (const float*)d_in[13];
  float* out = (float*)d_out;

  // ---- workspace layout (floats) ----
  float* ws   = (float*)d_ws;
  float* q    = ws;  ws += (size_t)BATCH * H_DIM;
  float* span = ws;  ws += (size_t)BATCH * H_DIM;
  float* syn  = ws;  ws += (size_t)BATCH * H_DIM;
  float* deg  = ws;  ws += (size_t)BATCH * S_LEN;
  float* buf1 = ws;  ws += (size_t)BATCH * S_LEN * H_DIM;  // xw (GEMM out)
  float* buf2 = ws;                                        // conv out / next GEMM in

  const int M     = BATCH * S_LEN;
  const int total = BATCH * S_LEN * H_DIM;
  const dim3 gemmGrid(M / MT, (H_DIM + NT - 1) / NT);   // 1024 x 2
  const dim3 scatGrid(NEDGE / 8, BATCH);                // 256 x 128
  const int  ewGrid = (total + 255) / 256;

  // attention-span branch
  aspectq_kernel<<<BATCH, 256, 0, stream>>>(h, ap, W_span, b_span, q);
  span_kernel   <<<BATCH, 256, 0, stream>>>(h, q, span);

  // GCN branch
  deg_kernel<<<BATCH, 256, 0, stream>>>(ei, ew, deg);

  // layer 1: xw1 = h @ W1 ; out = scatter + self + b1 ; relu
  gemm_bf16wmma_kernel<<<gemmGrid, 256, 0, stream>>>(h, W1, buf1);
  combine_init_kernel <<<ewGrid,   256, 0, stream>>>(buf1, deg, b1, buf2, total);
  scatter_kernel      <<<scatGrid, 256, 0, stream>>>(buf1, deg, ei, ew, buf2);
  relu_kernel         <<<ewGrid,   256, 0, stream>>>(buf2, total);

  // layer 2: xw2 = x1 @ W2 ; out = scatter + self + b2
  gemm_bf16wmma_kernel<<<gemmGrid, 256, 0, stream>>>(buf2, W2, buf1);
  combine_init_kernel <<<ewGrid,   256, 0, stream>>>(buf1, deg, b2, buf2, total);
  scatter_kernel      <<<scatGrid, 256, 0, stream>>>(buf1, deg, ei, ew, buf2);

  // heads
  syn_kernel  <<<BATCH, 256, 0, stream>>>(buf2, ap, syn);
  final_kernel<<<BATCH, 256, 0, stream>>>(span, syn, Wg, bg, Wc, bc, out);
}
